// CausalSelfAttention2_50929722196617
// MI455X (gfx1250) — matrix operand order repaired and checked by reading, not compiled, not run on hardware
//
#include <hip/hip_runtime.h>
#include <hip/hip_bf16.h>

typedef _Float16 v16h __attribute__((ext_vector_type(16)));
typedef _Float16 v8h  __attribute__((ext_vector_type(8)));
typedef float    v8f  __attribute__((ext_vector_type(8)));

#define DEV __device__ __forceinline__

constexpr int kB  = 8;
constexpr int kT  = 8192;
constexpr int kC  = 128;
constexpr int kH  = 4;
constexpr int kHS = 32;
constexpr int kNG = 8;
constexpr int kG  = 1024;   // tokens per group (before mean token)
constexpr int kL  = 1025;   // group length incl. mean token
constexpr int kLP = 1056;   // padded group length (33 * 32)
constexpr int kTN = 8200;   // final token count per batch
constexpr int kUnits = kB * kH * kNG;   // 256 (b,h,g) units
constexpr float kQScale = 0.17677669529663688f;  // 1/sqrt(32)

// D = A(16x32 f16) * B(32x16 f16) + C(16x16 f32)
DEV v8f wmma16(v16h a, v16h b, v8f c) {
  return __builtin_amdgcn_wmma_f32_16x16x32_f16(
      /*neg_a=*/false, a, /*neg_b=*/false, b,
      /*c_mod=*/(short)0, c, /*reuse_a=*/false, /*reuse_b=*/false);
}

// A-operand (16x32, f16): lane = row (l&15); hi=l>>4 selects K-halves.
// elems 0..7  = K in [16*hi, 16*hi+8)
// elems 8..15 = K in [16*hi+8, 16*hi+16)
DEV v16h a_from_h(const _Float16* row, int hi) {
  v8h lo = *(const v8h*)(row + 16 * hi);
  v8h hh = *(const v8h*)(row + 16 * hi + 8);
  return __builtin_shufflevector(lo, hh, 0,1,2,3,4,5,6,7,8,9,10,11,12,13,14,15);
}
DEV v16h a_from_f(const float* row, int hi) {
  v16h r;
#pragma unroll
  for (int i = 0; i < 8; i++) {
    r[i]     = (_Float16)row[16 * hi + i];
    r[8 + i] = (_Float16)row[16 * hi + 8 + i];
  }
  return r;
}
// B-operand column chunk: 16 contiguous values (K range [16*hi, 16*hi+16))
DEV v16h b_from_f(const float* p) {
  v16h r;
#pragma unroll
  for (int i = 0; i < 16; i++) r[i] = (_Float16)p[i];
  return r;
}

// ---------------- Kernel 1: qkv = x @ w_attn.T, scatter to grouped layout ----
__global__ __launch_bounds__(32)
void qkv_gemm(const float* __restrict__ x, const float* __restrict__ w_attn,
              _Float16* __restrict__ Q, _Float16* __restrict__ K,
              _Float16* __restrict__ Vt) {
  const int blk = blockIdx.x;
  const int nt = blk % 24;            // N tile (384 / 16)
  const int mt = (blk / 24) % 512;    // M tile within batch (8192 / 16)
  const int b  = blk / (24 * 512);
  const int lane = threadIdx.x, nl = lane & 15, hi = lane >> 4;

  const float* xrow = x + ((size_t)b * kT + mt * 16 + nl) * kC;
  const float* wrow = w_attn + (size_t)(nt * 16 + nl) * kC;

  v8f acc = {};
#pragma unroll
  for (int ks = 0; ks < 4; ks++) {
    v16h a = a_from_f(xrow + 32 * ks, hi);
    v16h w = b_from_f(wrow + 32 * ks + 16 * hi);
    acc = wmma16(a, w, acc);
  }

  const int ng = nt * 16 + nl;       // output column 0..383
  const int which = ng >> 7;         // 0=q 1=k 2=v
  const int c = ng & 127;
  const int h = c >> 5;
  const int d = c & 31;
#pragma unroll
  for (int r = 0; r < 8; r++) {
    const int t = mt * 16 + r + 8 * hi;      // C-layout: rows r / r+8
    const int g = t >> 10, i = t & 1023;
    const size_t u = (size_t)((b * kH + h) * kNG + g);
    const float val = acc[r];
    if (which == 0)
      Q[u * kLP * kHS + (size_t)i * kHS + d] = (_Float16)(val * kQScale);
    else if (which == 1)
      K[u * kLP * kHS + (size_t)i * kHS + d] = (_Float16)val;
    else
      Vt[(u * kHS + d) * kLP + i] = (_Float16)val;   // dim-major V
  }
}

// ---------------- Kernel 2: append mean token, zero padding ----------------
__global__ __launch_bounds__(256)
void mean_pad(_Float16* __restrict__ Q, _Float16* __restrict__ K,
              _Float16* __restrict__ Vt) {
  const int idx = blockIdx.x * 256 + threadIdx.x;   // < 3*8192
  const int tsr = idx >> 13;
  const int rem = idx & 8191;
  const int u = rem >> 5, d = rem & 31;
  if (tsr < 2) {
    _Float16* base = (tsr == 0 ? Q : K) + (size_t)u * kLP * kHS + d;
    float s = 0.f;
    for (int i = 0; i < kG; i++) s += (float)base[(size_t)i * kHS];
    base[(size_t)kG * kHS] = (_Float16)(s * (1.f / kG));
    for (int i = kG + 1; i < kLP; i++) base[(size_t)i * kHS] = (_Float16)0.f;
  } else {
    _Float16* base = Vt + ((size_t)u * kHS + d) * kLP;
    float s = 0.f;
    for (int i = 0; i < kG; i++) s += (float)base[i];
    base[kG] = (_Float16)(s * (1.f / kG));
    for (int i = kG + 1; i < kLP; i++) base[i] = (_Float16)0.f;
  }
}

// ---------------- Kernel 3: causal flash attention per (b,h,g) -------------
__global__ __launch_bounds__(32)
void group_attn(const _Float16* __restrict__ Q, const _Float16* __restrict__ K,
                const _Float16* __restrict__ Vt, _Float16* __restrict__ XA) {
  __shared__ _Float16 plds[16 * 32];
  const int qt = blockIdx.x % 65;          // q tile (0..64), tile 64 = mean row
  const int u  = blockIdx.x / 65;
  const int lane = threadIdx.x, nl = lane & 15, hi = lane >> 4;

  const _Float16* q  = Q  + (size_t)u * kLP * kHS;
  const _Float16* k  = K  + (size_t)u * kLP * kHS;
  const _Float16* v  = Vt + (size_t)u * kHS * kLP;
  _Float16*       xa = XA + (size_t)u * kL * kHS;

  const int qi0 = qt * 16;
  const v16h aq = a_from_h(q + (size_t)(qi0 + nl) * kHS, hi);  // Q pre-scaled

  v8f O0 = {}, O1 = {};
  float m[8], s[8];
#pragma unroll
  for (int r = 0; r < 8; r++) { m[r] = -1e30f; s[r] = 0.f; }

  const int nkb = (qi0 + 47) >> 5;   // 32-key blocks covering j <= qi0+15 (and mean key)
  for (int kb32 = 0; kb32 < nkb; kb32++) {
    const int kb = kb32 * 32;
    // S tiles: B column n = key token (kb+n) features, contiguous 16 f16
    const v16h bk0 = *(const v16h*)(k + (size_t)(kb + nl) * kHS + 16 * hi);
    const v16h bk1 = *(const v16h*)(k + (size_t)(kb + 16 + nl) * kHS + 16 * hi);
    const v8f zero = {};
    v8f S0 = wmma16(aq, bk0, zero);
    v8f S1 = wmma16(aq, bk1, zero);

    const int j0 = kb + nl;
    const int j1 = kb + 16 + nl;
#pragma unroll
    for (int r = 0; r < 8; r++) {
      const int i = qi0 + r + 8 * hi;        // global query row for this reg/half
      float x0 = ((j0 <= i) && (j0 <= kG)) ? S0[r] : -1e30f;
      float x1 = ((j1 <= i) && (j1 <= kG)) ? S1[r] : -1e30f;
      float mx = fmaxf(x0, x1);
      mx = fmaxf(mx, __shfl_xor(mx, 1));
      mx = fmaxf(mx, __shfl_xor(mx, 2));
      mx = fmaxf(mx, __shfl_xor(mx, 4));
      mx = fmaxf(mx, __shfl_xor(mx, 8));     // row max within 16-lane half
      const float mn   = fmaxf(m[r], mx);
      const float corr = __expf(m[r] - mn);
      const float p0 = __expf(x0 - mn);      // masked -> exp(-1e30) == 0
      const float p1 = __expf(x1 - mn);
      float rs = p0 + p1;
      rs += __shfl_xor(rs, 1);
      rs += __shfl_xor(rs, 2);
      rs += __shfl_xor(rs, 4);
      rs += __shfl_xor(rs, 8);
      s[r] = s[r] * corr + rs;
      m[r] = mn;
      O0[r] *= corr; O1[r] *= corr;
      plds[(r + 8 * hi) * 32 + nl]      = (_Float16)p0;   // re-layout P via LDS
      plds[(r + 8 * hi) * 32 + 16 + nl] = (_Float16)p1;
    }
    const v16h ap  = a_from_h(plds + (size_t)nl * 32, hi);       // P as A-operand
    const v16h bv0 = *(const v16h*)(v + (size_t)nl * kLP + kb + 16 * hi);
    const v16h bv1 = *(const v16h*)(v + (size_t)(16 + nl) * kLP + kb + 16 * hi);
    O0 = wmma16(ap, bv0, O0);   // dims 0..15
    O1 = wmma16(ap, bv1, O1);   // dims 16..31
  }

#pragma unroll
  for (int r = 0; r < 8; r++) {
    const int i = qi0 + r + 8 * hi;
    if (i <= kG) {
      const float inv = 1.f / s[r];
      xa[(size_t)i * kHS + nl]      = (_Float16)(O0[r] * inv);
      xa[(size_t)i * kHS + 16 + nl] = (_Float16)(O1[r] * inv);
    }
  }
}

// ---------------- Kernel 4: 8-token causal attention over group means ------
__global__ __launch_bounds__(256)
void summary_attn(const _Float16* __restrict__ Q, const _Float16* __restrict__ K,
                  const _Float16* __restrict__ XA, _Float16* __restrict__ Y) {
  const int t = threadIdx.x;           // 0..255 == (b,h,g)
  const int b = t >> 5, h = (t >> 3) & 3, g = t & 7;
  const int ub = (b * kH + h) * kNG;
  float qv[kHS];
#pragma unroll
  for (int d = 0; d < kHS; d++)
    qv[d] = (float)Q[((size_t)(ub + g) * kLP + kG) * kHS + d];  // scaled q mean
  float sc[kNG];
  float mx = -1e30f;
  for (int j = 0; j <= g; j++) {
    const _Float16* kr = K + ((size_t)(ub + j) * kLP + kG) * kHS;
    float dot = 0.f;
    for (int d = 0; d < kHS; d++) dot += qv[d] * (float)kr[d];
    sc[j] = dot;
    mx = fmaxf(mx, dot);
  }
  float ssum = 0.f;
  for (int j = 0; j <= g; j++) { sc[j] = __expf(sc[j] - mx); ssum += sc[j]; }
  const float inv = 1.f / ssum;
  for (int d = 0; d < kHS; d++) {
    float acc = 0.f;
    for (int j = 0; j <= g; j++)
      acc += sc[j] * (float)XA[((size_t)(ub + j) * kL + kG) * kHS + d];
    Y[(size_t)(ub + g) * kHS + d] = (_Float16)(acc * inv);
  }
}

// ---------------- Kernel 5: gather rows + out = xa @ w_proj.T --------------
__global__ __launch_bounds__(32)
void out_proj(const _Float16* __restrict__ XA, const _Float16* __restrict__ Y,
              const float* __restrict__ w_proj, float* __restrict__ out) {
  const int blk = blockIdx.x;
  const int nt = blk & 7;                 // N tile (128/16)
  const int mt = (blk >> 3) % 513;        // M tile (8208/16 covers 8200)
  const int b  = blk / (8 * 513);
  const int lane = threadIdx.x, nl = lane & 15, hi = lane >> 4;

  int tt = mt * 16 + nl;
  if (tt > kTN - 1) tt = kTN - 1;         // clamp padded rows (not stored)
  const int g = tt / kL, rr = tt % kL;

  const int ng = nt * 16 + nl;
  const float* wrow = w_proj + (size_t)ng * kC;

  v8f acc = {};
#pragma unroll
  for (int h = 0; h < 4; h++) {           // k-step == head (32 dims each)
    const _Float16* arow;
    if (g == 0)
      arow = XA + ((size_t)((b * kH + h) * kNG + 0) * kL + rr) * kHS;
    else if (rr == 0)
      arow = Y + (size_t)((b * kH + h) * kNG + (g - 1)) * kHS;
    else
      arow = XA + ((size_t)((b * kH + h) * kNG + g) * kL + (rr - 1)) * kHS;
    v16h a = a_from_h(arow, hi);
    v16h w = b_from_f(wrow + 32 * h + 16 * hi);
    acc = wmma16(a, w, acc);
  }
#pragma unroll
  for (int r = 0; r < 8; r++) {
    const int trow = mt * 16 + r + 8 * hi;
    if (trow < kTN)
      out[((size_t)b * kTN + trow) * kC + ng] = acc[r];
  }
}

// ---------------------------------------------------------------------------
extern "C" void kernel_launch(void* const* d_in, const int* in_sizes, int n_in,
                              void* d_out, int out_size, void* d_ws, size_t ws_size,
                              hipStream_t stream) {
  (void)in_sizes; (void)n_in; (void)out_size; (void)ws_size;
  const float* x      = (const float*)d_in[0];
  const float* w_attn = (const float*)d_in[1];
  const float* w_proj = (const float*)d_in[2];
  float* out = (float*)d_out;

  _Float16* ws = (_Float16*)d_ws;
  const size_t SZQ  = (size_t)kUnits * kLP * kHS;   // 8,650,752 f16
  const size_t SZXA = (size_t)kUnits * kL * kHS;    // 8,396,800 f16
  _Float16* Q  = ws;
  _Float16* K  = Q + SZQ;
  _Float16* Vt = K + SZQ;
  _Float16* XA = Vt + SZQ;
  _Float16* Y  = XA + SZXA;                          // total ~65.5 MB

  qkv_gemm   <<<dim3(kB * 512 * 24), dim3(32),  0, stream>>>(x, w_attn, Q, K, Vt);
  mean_pad   <<<dim3(96),            dim3(256), 0, stream>>>(Q, K, Vt);
  group_attn <<<dim3(kUnits * 65),   dim3(32),  0, stream>>>(Q, K, Vt, XA);
  summary_attn<<<dim3(1),            dim3(256), 0, stream>>>(Q, K, XA, Y);
  out_proj   <<<dim3(kB * 513 * 8),  dim3(32),  0, stream>>>(XA, Y, w_proj, out);
}
